// SelfAttention_32804960207443
// MI455X (gfx1250) — compile-verified, matrix operand order checked by reading, not run on hardware
//
#include <hip/hip_runtime.h>
#include <hip/hip_bf16.h>

// ---------------------------------------------------------------------------
// CDNA5 (gfx1250, wave32) self-attention using v_wmma_f32_16x16x32_bf16.
// ---------------------------------------------------------------------------

typedef __attribute__((ext_vector_type(16))) __bf16 v16bf;
typedef __attribute__((ext_vector_type(8)))  __bf16 v8bf;
typedef __attribute__((ext_vector_type(8)))  float  v8f;
typedef __attribute__((ext_vector_type(4)))  float  v4f;

#define B_  2
#define S_  2048
#define D_  1024
#define H_  16
#define DH_ 64
#define P_  1024

// ---- fragment loaders -----------------------------------------------------
// 16-bit A/B 16x32 VGPR layout (ISA 7.12.2): lane = (g,m), g=lane>>4, m=lane&15.
// Fragment elems [0..7]  = K = g*8 + 0..7
// Fragment elems [8..15] = K = 16 + g*8 + 0..7
// -> two contiguous 16-byte loads per lane when the source row is contiguous.

__device__ __forceinline__ v16bf load_frag(const __bf16* __restrict__ p, int g) {
    v8bf lo = *(const v8bf*)(p + g * 8);
    v8bf hi = *(const v8bf*)(p + 16 + g * 8);
    v16bf f;
#pragma unroll
    for (int i = 0; i < 8; ++i) { f[i] = lo[i]; f[i + 8] = hi[i]; }
    return f;
}

__device__ __forceinline__ v16bf load_frag(const float* __restrict__ p, int g) {
    v4f l0 = *(const v4f*)(p + g * 8);
    v4f l1 = *(const v4f*)(p + g * 8 + 4);
    v4f h0 = *(const v4f*)(p + 16 + g * 8);
    v4f h1 = *(const v4f*)(p + 16 + g * 8 + 4);
    v16bf f;
#pragma unroll
    for (int i = 0; i < 4; ++i) {
        f[i]      = (__bf16)l0[i];
        f[4 + i]  = (__bf16)l1[i];
        f[8 + i]  = (__bf16)h0[i];
        f[12 + i] = (__bf16)h1[i];
    }
    return f;
}

// ---- generic batched WMMA GEMM:  C = alpha * A * B^T  ---------------------
// A: [M,K] row-major (bf16 or fp32->bf16 on the fly), lda
// Bt: [N,K] row-major bf16 (i.e. B transposed), ldb
// C: [M,N], ldc  (fp32 or bf16); optional column mask (-1e12), optional
// per-head-transposed V store.
// grid.z batches heads: z -> (zb = z/Hsplit, zh = z%Hsplit), base offsets
// off = zb*OffB + zh*OffH for each operand.
// Block = 256 threads = 8 waves; wave computes a 16x64 strip (4 WMMA tiles,
// A-fragment reused 4x); block tile = 128x64.

template <typename AT, typename CT>
__global__ void __launch_bounds__(256)
gemm_wmma_bf16(const AT* __restrict__ A, const __bf16* __restrict__ Bt,
               CT* __restrict__ C,
               int M, int N, int Kdim,
               long long lda, long long ldb, long long ldc,
               long long aOffB, long long aOffH,
               long long bOffB, long long bOffH,
               long long cOffB, long long cOffH,
               int Hsplit, float alpha,
               const int* __restrict__ mask, long long maskStride,
               int storeVT)
{
    const int z  = blockIdx.z;
    const int zb = z / Hsplit;
    const int zh = z % Hsplit;

    const AT*     Az = A  + (long long)zb * aOffB + (long long)zh * aOffH;
    const __bf16* Bz = Bt + (long long)zb * bOffB + (long long)zh * bOffH;
    CT*           Cz = C  + (long long)zb * cOffB + (long long)zh * cOffH;
    const int* maskRow = mask ? (mask + (long long)zb * maskStride) : nullptr;

    const int tid  = threadIdx.x;
    const int wave = tid >> 5;
    const int lane = tid & 31;
    const int g    = lane >> 4;   // half-wave group (K-split)
    const int ln   = lane & 15;   // row (A) / col (B,C) within the tile

    const int rowBase = blockIdx.y * 128 + wave * 16;
    const int colBase = blockIdx.x * 64;
    if (rowBase >= M) return;     // wave-uniform; EXEC stays all-ones for WMMA

    v8f acc0 = {}, acc1 = {}, acc2 = {}, acc3 = {};

    const AT*     aRow  = Az + (long long)(rowBase + ln) * lda;
    const __bf16* bRow0 = Bz + (long long)(colBase +  0 + ln) * ldb;
    const __bf16* bRow1 = Bz + (long long)(colBase + 16 + ln) * ldb;
    const __bf16* bRow2 = Bz + (long long)(colBase + 32 + ln) * ldb;
    const __bf16* bRow3 = Bz + (long long)(colBase + 48 + ln) * ldb;

    for (int k = 0; k < Kdim; k += 32) {
        v16bf af = load_frag(aRow  + k, g);
        v16bf b0 = load_frag(bRow0 + k, g);
        v16bf b1 = load_frag(bRow1 + k, g);
        v16bf b2 = load_frag(bRow2 + k, g);
        v16bf b3 = load_frag(bRow3 + k, g);
        acc0 = __builtin_amdgcn_wmma_f32_16x16x32_bf16(false, af, false, b0, (short)0, acc0, false, false);
        acc1 = __builtin_amdgcn_wmma_f32_16x16x32_bf16(false, af, false, b1, (short)0, acc1, false, false);
        acc2 = __builtin_amdgcn_wmma_f32_16x16x32_bf16(false, af, false, b2, (short)0, acc2, false, false);
        acc3 = __builtin_amdgcn_wmma_f32_16x16x32_bf16(false, af, false, b3, (short)0, acc3, false, false);
    }

    // C/D layout (ISA 7.12.2): lane (g, ln) VGPR r holds C[rowBase + g*8 + r][colBase + t*16 + ln]
    v8f accs[4] = {acc0, acc1, acc2, acc3};
#pragma unroll
    for (int t = 0; t < 4; ++t) {
        const int col = colBase + t * 16 + ln;
        const bool masked = maskRow && (maskRow[col] == 0);
#pragma unroll
        for (int r = 0; r < 8; ++r) {
            const int row = rowBase + g * 8 + r;
            float v = accs[t][r] * alpha;
            if (masked) v = -1e12f;
            if (!storeVT) {
                Cz[(long long)row * ldc + col] = (CT)v;
            } else {
                // V projection stored per-head transposed: Vt[((b*H+h)*DH+dh)*S + s]
                const int vb = row / S_,  vs  = row % S_;
                const int vh = col / DH_, vdh = col % DH_;
                Cz[(((long long)(vb * H_ + vh)) * DH_ + vdh) * S_ + vs] = (CT)v;
            }
        }
    }
}

// ---- precision conversion helpers -----------------------------------------

__global__ void cvt_f32_to_bf16(const float* __restrict__ in, __bf16* __restrict__ out,
                                long long n) {
    long long i = (long long)blockIdx.x * blockDim.x + threadIdx.x;
    if (i < n) out[i] = (__bf16)in[i];
}

// out[c*R + r] = in[r*Ccols + c]   (transpose + convert; coalesced writes)
__global__ void transpose_cvt(const float* __restrict__ in, __bf16* __restrict__ out,
                              int R, int Ccols) {
    long long i = (long long)blockIdx.x * blockDim.x + threadIdx.x;
    long long n = (long long)R * Ccols;
    if (i >= n) return;
    int c = (int)(i / R);
    int r = (int)(i % R);
    out[i] = (__bf16)in[(long long)r * Ccols + c];
}

// ---- softmax over rows of length 2048, row fully register-resident --------
// One block per row: exactly 1 read + 1 write of the 537MB attn tensor.

__global__ void __launch_bounds__(256)
softmax_rows_2048(float* __restrict__ attn) {
    __shared__ float sred[256];
    const long long base = (long long)blockIdx.x * S_;
    const int tid = threadIdx.x;

    float vals[8];
#pragma unroll
    for (int j = 0; j < 8; ++j) vals[j] = attn[base + tid + j * 256];

    float m = vals[0];
#pragma unroll
    for (int j = 1; j < 8; ++j) m = fmaxf(m, vals[j]);
    sred[tid] = m;
    __syncthreads();
    for (int s = 128; s > 0; s >>= 1) {
        if (tid < s) sred[tid] = fmaxf(sred[tid], sred[tid + s]);
        __syncthreads();
    }
    m = sred[0];
    __syncthreads();

    float sum = 0.f;
#pragma unroll
    for (int j = 0; j < 8; ++j) { vals[j] = __expf(vals[j] - m); sum += vals[j]; }
    sred[tid] = sum;
    __syncthreads();
    for (int s = 128; s > 0; s >>= 1) {
        if (tid < s) sred[tid] += sred[tid + s];
        __syncthreads();
    }
    const float inv = 1.0f / sred[0];
#pragma unroll
    for (int j = 0; j < 8; ++j) attn[base + tid + j * 256] = vals[j] * inv;
}

// ---------------------------------------------------------------------------

extern "C" void kernel_launch(void* const* d_in, const int* in_sizes, int n_in,
                              void* d_out, int out_size, void* d_ws, size_t ws_size,
                              hipStream_t stream)
{
    (void)in_sizes; (void)n_in; (void)out_size; (void)ws_size;

    const float* x   = (const float*)d_in[0];
    const int*   msk = (const int*)d_in[1];   // [B,1,1,S] int32
    const float* Wq  = (const float*)d_in[2];
    const float* Wk  = (const float*)d_in[3];
    const float* Wv  = (const float*)d_in[4];
    const float* Wo  = (const float*)d_in[5];

    const long long BSD = (long long)B_ * S_ * D_;   // 4,194,304
    const long long BSP = (long long)B_ * S_ * P_;   // 4,194,304
    const long long DP  = (long long)D_ * P_;
    const long long PP  = (long long)P_ * P_;

    // workspace layout (bf16), ~50MB total
    __bf16* xbf = (__bf16*)d_ws;
    __bf16* Wqt = xbf + BSD;
    __bf16* Wkt = Wqt + DP;
    __bf16* Wvt = Wkt + DP;
    __bf16* Wot = Wvt + DP;
    __bf16* Qb  = Wot + PP;     // [B*S, P]
    __bf16* Kb  = Qb  + BSP;    // [B*S, P]
    __bf16* Vt  = Kb  + BSP;    // [B*H*DH, S] (per-head transposed V)
    __bf16* AV  = Vt  + BSP;    // [B*S, P] attn@V result

    float* out  = (float*)d_out;        // [B,S,P]
    float* attn = out + BSP;            // [B,H,S,S]

    // 1) convert x, transpose+convert weights (Bt form for contiguous frags)
    cvt_f32_to_bf16<<<dim3((unsigned)((BSD + 255) / 256)), dim3(256), 0, stream>>>(x, xbf, BSD);
    transpose_cvt<<<dim3((unsigned)((DP + 255) / 256)), dim3(256), 0, stream>>>(Wq, Wqt, D_, P_);
    transpose_cvt<<<dim3((unsigned)((DP + 255) / 256)), dim3(256), 0, stream>>>(Wk, Wkt, D_, P_);
    transpose_cvt<<<dim3((unsigned)((DP + 255) / 256)), dim3(256), 0, stream>>>(Wv, Wvt, D_, P_);
    transpose_cvt<<<dim3((unsigned)((PP + 255) / 256)), dim3(256), 0, stream>>>(Wo, Wot, P_, P_);

    const dim3 blk(256);

    // 2) projections: [4096,1024] x [1024,1024]
    {
        dim3 grd(P_ / 64, (B_ * S_) / 128, 1);
        gemm_wmma_bf16<__bf16, __bf16><<<grd, blk, 0, stream>>>(
            xbf, Wqt, Qb, B_ * S_, P_, D_, D_, D_, P_,
            0, 0, 0, 0, 0, 0, 1, 1.0f, nullptr, 0, 0);
        gemm_wmma_bf16<__bf16, __bf16><<<grd, blk, 0, stream>>>(
            xbf, Wkt, Kb, B_ * S_, P_, D_, D_, D_, P_,
            0, 0, 0, 0, 0, 0, 1, 1.0f, nullptr, 0, 0);
        gemm_wmma_bf16<__bf16, __bf16><<<grd, blk, 0, stream>>>(
            xbf, Wvt, Vt, B_ * S_, P_, D_, D_, D_, P_,
            0, 0, 0, 0, 0, 0, 1, 1.0f, nullptr, 0, /*storeVT=*/1);
    }

    // 3) logits = Q K^T / sqrt(DH), masked, fp32 -> attn region of d_out
    {
        dim3 grd(S_ / 64, S_ / 128, B_ * H_);
        gemm_wmma_bf16<__bf16, float><<<grd, blk, 0, stream>>>(
            Qb, Kb, attn, S_, S_, DH_, P_, P_, S_,
            /*aOffB=*/(long long)S_ * P_, /*aOffH=*/DH_,
            /*bOffB=*/(long long)S_ * P_, /*bOffH=*/DH_,
            /*cOffB=*/(long long)H_ * S_ * S_, /*cOffH=*/(long long)S_ * S_,
            H_, 0.125f, msk, S_, 0);
    }

    // 4) softmax over each of the B*H*S rows (register-resident rows)
    softmax_rows_2048<<<dim3(B_ * H_ * S_), blk, 0, stream>>>(attn);

    // 5) attn @ V : per head [2048,2048] x [2048,64]; A is fp32 attn
    {
        dim3 grd(DH_ / 64, S_ / 128, B_ * H_);
        gemm_wmma_bf16<float, __bf16><<<grd, blk, 0, stream>>>(
            attn, Vt, AV, S_, DH_, S_, S_, S_, P_,
            /*aOffB=*/(long long)H_ * S_ * S_, /*aOffH=*/(long long)S_ * S_,
            /*bOffB=*/(long long)H_ * DH_ * S_, /*bOffH=*/(long long)DH_ * S_,
            /*cOffB=*/(long long)S_ * P_, /*cOffH=*/DH_,
            H_, 1.0f, nullptr, 0, 0);
    }

    // 6) output projection: out = AV @ Wo, fp32 -> d_out
    {
        dim3 grd(P_ / 64, (B_ * S_) / 128, 1);
        gemm_wmma_bf16<__bf16, float><<<grd, blk, 0, stream>>>(
            AV, Wot, out, B_ * S_, P_, P_, P_, P_, P_,
            0, 0, 0, 0, 0, 0, 1, 1.0f, nullptr, 0, 0);
    }
}
